// SchnetNet_29944511988252
// MI455X (gfx1250) — matrix operand order repaired and checked by reading, not compiled, not run on hardware
//
#include <hip/hip_runtime.h>
#include <hip/hip_bf16.h>

#define F 128
#define L 3
#define N_RBF 300
#define NRBF_PAD 320
#define B 64
#define NATOM 128
#define P 4096

typedef __attribute__((ext_vector_type(16))) __bf16 v16bf;
typedef __attribute__((ext_vector_type(8)))  float  v8f;

union Frag { unsigned u[8]; v16bf v; };

__device__ __forceinline__ unsigned pack_bf16(float lo, float hi) {
    unsigned a = __builtin_bit_cast(unsigned, lo);
    unsigned b = __builtin_bit_cast(unsigned, hi);
    a = (a + 0x7fffu + ((a >> 16) & 1u)) >> 16;
    b = (b + 0x7fffu + ((b >> 16) & 1u)) >> 16;
    return a | (b << 16);
}
__device__ __forceinline__ unsigned short f2bf(float x) {
    unsigned a = __builtin_bit_cast(unsigned, x);
    return (unsigned short)((a + 0x7fffu + ((a >> 16) & 1u)) >> 16);
}
// ShiftedSoftPlus: softplus(x) - ln2, numerically stable
__device__ __forceinline__ float sspf(float x) {
    float e = __expf(-fabsf(x));
    return fmaxf(x, 0.0f) + log1pf(e) - 0.69314718056f;
}

// ---------------------------------------------------------------------------
// WMMA GEMM, row-strip version: one wave computes a 16 x (16*NTILES) strip.
// A K-fragments are loaded+converted ONCE and reused across all N-tiles.
// C = op(A[M,K] @ W[K,N] + bias); W pre-transposed bf16 [N,K].
// ACT: 0=none 1=ssp.  RES: 1 -> C += result.
// ---------------------------------------------------------------------------
template<int K, int NTILES, int ACT, int RES>
__global__ void __launch_bounds__(32)
gemm_bf16_kernel(const float* __restrict__ A,             // [M,K] f32
                 const unsigned short* __restrict__ Wt,   // [N,K] bf16 (transposed)
                 const float* __restrict__ bias,          // [N] or null
                 float* __restrict__ C)                   // [M, 16*NTILES]
{
    constexpr int NKB = K / 32;
    const int lane  = threadIdx.x & 31;
    const int m0    = blockIdx.x * 16;
    const int mrow  = lane & 15;
    const int khalf = lane >> 4;
    const int ncol0 = lane & 15;

    // Preload + convert all A fragments for this row strip (reused 8x)
    Frag fa[NKB];
    const float* Arow = A + (size_t)(m0 + mrow) * K;
#pragma unroll
    for (int kb = 0; kb < NKB; ++kb) {
#pragma unroll
        for (int v = 0; v < 8; ++v) {
            int ka = kb * 32 + ((v >> 2) << 4) + (khalf << 3) + ((v & 3) << 1);
            float2 p = *(const float2*)(Arow + ka);
            fa[kb].u[v] = pack_bf16(p.x, p.y);
        }
    }

#pragma unroll
    for (int nt = 0; nt < NTILES; ++nt) {
        v8f acc = {};
        const unsigned short* Wrow = Wt + (size_t)(nt * 16 + ncol0) * K;
#pragma unroll
        for (int kb = 0; kb < NKB; ++kb) {
            Frag fb;
#pragma unroll
            for (int v = 0; v < 8; ++v) {
                int kc = kb * 32 + ((v >> 2) << 4) + (khalf << 3) + ((v & 3) << 1);
                fb.u[v] = *(const unsigned*)(Wrow + kc);
            }
            acc = __builtin_amdgcn_wmma_f32_16x16x32_bf16(false, fa[kb].v, false, fb.v,
                                                          (short)0, acc, false, false);
        }
        const int ncol = nt * 16 + ncol0;
        float bv = bias ? bias[ncol] : 0.0f;
#pragma unroll
        for (int r = 0; r < 8; ++r) {
            int mm = m0 + r + (khalf << 3);
            float v = acc[r] + bv;
            if (ACT == 1) v = sspf(v);
            float* cp = C + (size_t)mm * (16 * NTILES) + ncol;
            if (RES) *cp += v; else *cp = v;
        }
    }
}

// ---------------------------------------------------------------------------
// Fused cfconv: one wave handles 32 neighbor pairs (two 16-row M-tiles) so
// every B-fragment (filter weights) is shared across two WMMAs.
//   rbf tile (32x320, on the fly) -> LDS bf16
//   GEMM1: hidden = ssp(rbf @ W1 + b1) -> LDS bf16 (32x128)
//   GEMM2: Wf = hidden @ W2 + b2
//   scatter: agg[b, idx_j[p], f] += Wf[p,f] * x_in[b, idx_j[p], f]
// Single-wave workgroup: __syncthreads() separates the LDS write and read stages.
// ---------------------------------------------------------------------------
__global__ void __launch_bounds__(32)
cfconv_kernel(const float* __restrict__ dists,          // [B,P]
              const int*   __restrict__ idx_j,          // [B,P]
              const float* __restrict__ x_in,           // [B,N,F]
              const unsigned short* __restrict__ W1t,   // [F, NRBF_PAD] bf16 (layer)
              const float* __restrict__ b1,             // [F]
              const unsigned short* __restrict__ W2t,   // [F, F] bf16
              const float* __restrict__ b2,             // [F]
              float* __restrict__ agg)                  // [B,N,F]
{
    __shared__ unsigned       rbf_pairs[32 * (NRBF_PAD / 2)];  // 20 KB
    __shared__ unsigned short hid[32 * F];                      // 8 KB

    const int lane = threadIdx.x & 31;
    const int b    = blockIdx.y;
    const int p0   = blockIdx.x * 32;

    const float step  = 30.0f / (float)(N_RBF - 1);
    const float coeff = -0.5f / (step * step);

    // Stage 1: rbf tile into LDS (k >= 300 harmless: W1t zero-padded there)
    for (int m = 0; m < 32; ++m) {
        float dm = dists[(size_t)b * P + p0 + m];
        for (int kk = lane; kk < NRBF_PAD / 2; kk += 32) {
            float t0 = dm - (float)(2 * kk) * step;
            float t1 = dm - (float)(2 * kk + 1) * step;
            float e0 = __expf(coeff * t0 * t0);
            float e1 = __expf(coeff * t1 * t1);
            rbf_pairs[m * (NRBF_PAD / 2) + kk] = pack_bf16(e0, e1);
        }
    }
    __syncthreads();

    const int mrow  = lane & 15;
    const int khalf = lane >> 4;
    const int ncol0 = lane & 15;

    // Stage 2: GEMM1 over K=320 for both M-tiles, bias + ssp -> LDS bf16
    for (int nt = 0; nt < 8; ++nt) {
        v8f acc0 = {}, acc1 = {};
        const unsigned short* Wrow = W1t + (size_t)(nt * 16 + ncol0) * NRBF_PAD;
#pragma unroll 2
        for (int kb = 0; kb < NRBF_PAD; kb += 32) {
            Frag fa0, fa1, fb;
#pragma unroll
            for (int v = 0; v < 8; ++v) {
                int ka = kb + ((v >> 2) << 4) + (khalf << 3) + ((v & 3) << 1);
                fa0.u[v] = rbf_pairs[mrow * (NRBF_PAD / 2) + (ka >> 1)];
                fa1.u[v] = rbf_pairs[(16 + mrow) * (NRBF_PAD / 2) + (ka >> 1)];
                int kc = kb + ((v >> 2) << 4) + (khalf << 3) + ((v & 3) << 1);
                fb.u[v] = *(const unsigned*)(Wrow + kc);
            }
            acc0 = __builtin_amdgcn_wmma_f32_16x16x32_bf16(false, fa0.v, false, fb.v,
                                                           (short)0, acc0, false, false);
            acc1 = __builtin_amdgcn_wmma_f32_16x16x32_bf16(false, fa1.v, false, fb.v,
                                                           (short)0, acc1, false, false);
        }
        float bv = b1[nt * 16 + ncol0];
#pragma unroll
        for (int r = 0; r < 8; ++r) {
            int mm = r + (khalf << 3);
            hid[mm * F + nt * 16 + ncol0]        = f2bf(sspf(acc0[r] + bv));
            hid[(16 + mm) * F + nt * 16 + ncol0] = f2bf(sspf(acc1[r] + bv));
        }
    }
    __syncthreads();

    // Neighbor indices for the rows this lane owns in the C-layout
    int jj0[8], jj1[8];
#pragma unroll
    for (int r = 0; r < 8; ++r) {
        jj0[r] = idx_j[(size_t)b * P + p0 + (khalf << 3) + r];
        jj1[r] = idx_j[(size_t)b * P + p0 + 16 + (khalf << 3) + r];
    }

    // Stage 3: GEMM2 over K=128 for both M-tiles + fused gather-mul-scatter
    const unsigned* hidp = (const unsigned*)hid;
    for (int nt = 0; nt < 8; ++nt) {
        v8f acc0 = {}, acc1 = {};
        const unsigned short* Wrow = W2t + (size_t)(nt * 16 + ncol0) * F;
#pragma unroll
        for (int kb = 0; kb < F; kb += 32) {
            Frag fa0, fa1, fb;
#pragma unroll
            for (int v = 0; v < 8; ++v) {
                int ka = kb + ((v >> 2) << 4) + (khalf << 3) + ((v & 3) << 1);
                fa0.u[v] = hidp[mrow * (F / 2) + (ka >> 1)];
                fa1.u[v] = hidp[(16 + mrow) * (F / 2) + (ka >> 1)];
                int kc = kb + ((v >> 2) << 4) + (khalf << 3) + ((v & 3) << 1);
                fb.u[v] = *(const unsigned*)(Wrow + kc);
            }
            acc0 = __builtin_amdgcn_wmma_f32_16x16x32_bf16(false, fa0.v, false, fb.v,
                                                           (short)0, acc0, false, false);
            acc1 = __builtin_amdgcn_wmma_f32_16x16x32_bf16(false, fa1.v, false, fb.v,
                                                           (short)0, acc1, false, false);
        }
        int fcol = nt * 16 + ncol0;
        float bv = b2[fcol];
#pragma unroll
        for (int r = 0; r < 8; ++r) {
            size_t base0 = ((size_t)b * NATOM + jj0[r]) * F + fcol;
            size_t base1 = ((size_t)b * NATOM + jj1[r]) * F + fcol;
            float m0v = (acc0[r] + bv) * x_in[base0];
            float m1v = (acc1[r] + bv) * x_in[base1];
            __hip_atomic_fetch_add(&agg[base0], m0v,
                                   __ATOMIC_RELAXED, __HIP_MEMORY_SCOPE_AGENT);
            __hip_atomic_fetch_add(&agg[base1], m1v,
                                   __ATOMIC_RELAXED, __HIP_MEMORY_SCOPE_AGENT);
        }
    }
}

// ---------------------------------------------------------------------------
// Embedding gather: x[b,n,:] = emb[Z[b,n]]
// ---------------------------------------------------------------------------
__global__ void embed_kernel(const int* __restrict__ Z, const float* __restrict__ emb,
                             float* __restrict__ x) {
    int i = blockIdx.x * blockDim.x + threadIdx.x;   // over B*N*F
    int row = i >> 7;
    int f = i & (F - 1);
    x[i] = emb[Z[row] * F + f];
}

// ---------------------------------------------------------------------------
// Weight transpose + bf16 convert (+zero K-pad): dst[mat][n][k] = src[mat][k][n]
// ---------------------------------------------------------------------------
__global__ void wconv_kernel(const float* __restrict__ src, unsigned short* __restrict__ dst,
                             int K, int Ncols, int Kpad, int nmat) {
    int i = blockIdx.x * blockDim.x + threadIdx.x;
    if (i >= nmat * Ncols * Kpad) return;
    int k = i % Kpad;
    int t = i / Kpad;
    int n = t % Ncols;
    int mat = t / Ncols;
    float v = (k < K) ? src[((size_t)mat * K + k) * Ncols + n] : 0.0f;
    dst[i] = f2bf(v);
}

// ---------------------------------------------------------------------------
// Output head: per molecule, sum_n( ssp(x[n]@W1) @ W2 + b2 )
// ---------------------------------------------------------------------------
__global__ void __launch_bounds__(128)
head_kernel(const float* __restrict__ x, const float* __restrict__ w1,
            const float* __restrict__ w2, const float* __restrict__ b2,
            float* __restrict__ out) {
    __shared__ float sw1[F * 32];
    __shared__ float red[128];
    int tid = threadIdx.x;
    int b = blockIdx.x;
    for (int i = tid; i < F * 32; i += 128) sw1[i] = w1[i];
    __syncthreads();
    const float* xr = x + ((size_t)b * NATOM + tid) * F;
    float acc = 0.0f;
    for (int j = 0; j < 32; ++j) {
        float h = 0.0f;
#pragma unroll 8
        for (int f = 0; f < F; ++f) h += xr[f] * sw1[f * 32 + j];
        acc += sspf(h) * w2[j];
    }
    acc += b2[0];
    red[tid] = acc;
    __syncthreads();
    for (int s = 64; s > 0; s >>= 1) {
        if (tid < s) red[tid] += red[tid + s];
        __syncthreads();
    }
    if (tid == 0) out[b] = red[0];
}

// ---------------------------------------------------------------------------
extern "C" void kernel_launch(void* const* d_in, const int* in_sizes, int n_in,
                              void* d_out, int out_size, void* d_ws, size_t ws_size,
                              hipStream_t stream) {
    const int*   Z      = (const int*)  d_in[0];
    const float* dists  = (const float*)d_in[1];
    const int*   idxj   = (const int*)  d_in[2];
    const float* emb    = (const float*)d_in[3];
    const float* in2f_w = (const float*)d_in[4];
    const float* fw1    = (const float*)d_in[5];
    const float* fb1    = (const float*)d_in[6];
    const float* fw2    = (const float*)d_in[7];
    const float* fb2    = (const float*)d_in[8];
    const float* gw1    = (const float*)d_in[9];
    const float* gb1    = (const float*)d_in[10];
    const float* gw2    = (const float*)d_in[11];
    const float* gb2    = (const float*)d_in[12];
    const float* ow1    = (const float*)d_in[13];
    const float* ow2    = (const float*)d_in[14];
    const float* ob2    = (const float*)d_in[15];
    float* out = (float*)d_out;

    char* ws = (char*)d_ws;
    size_t off = 0;
    auto alloc = [&](size_t bytes) -> void* {
        void* p = ws + off;
        off = (off + bytes + 255) & ~(size_t)255;
        return p;
    };
    unsigned short* in2f_t = (unsigned short*)alloc((size_t)L * F * F * 2);
    unsigned short* w1t    = (unsigned short*)alloc((size_t)L * F * NRBF_PAD * 2);
    unsigned short* w2t    = (unsigned short*)alloc((size_t)L * F * F * 2);
    unsigned short* g1t    = (unsigned short*)alloc((size_t)L * F * F * 2);
    unsigned short* g2t    = (unsigned short*)alloc((size_t)L * F * F * 2);
    float* x    = (float*)alloc((size_t)B * NATOM * F * 4);
    float* x_in = (float*)alloc((size_t)B * NATOM * F * 4);
    float* agg  = (float*)alloc((size_t)B * NATOM * F * 4);
    float* tmp1 = (float*)alloc((size_t)B * NATOM * F * 4);

    // Pre-transpose + bf16-convert all weight matrices (tiny)
    int tF = L * F * F;
    wconv_kernel<<<(tF + 255) / 256, 256, 0, stream>>>(in2f_w, in2f_t, F, F, F, L);
    int tW1 = L * F * NRBF_PAD;
    wconv_kernel<<<(tW1 + 255) / 256, 256, 0, stream>>>(fw1, w1t, N_RBF, F, NRBF_PAD, L);
    wconv_kernel<<<(tF + 255) / 256, 256, 0, stream>>>(fw2, w2t, F, F, F, L);
    wconv_kernel<<<(tF + 255) / 256, 256, 0, stream>>>(gw1, g1t, F, F, F, L);
    wconv_kernel<<<(tF + 255) / 256, 256, 0, stream>>>(gw2, g2t, F, F, F, L);

    embed_kernel<<<(B * NATOM * F) / 256, 256, 0, stream>>>(Z, emb, x);

    const int Mrows = B * NATOM;           // 8192
    dim3 gemmGrid(Mrows / 16);             // 512 row-strips, 8 N-tiles each
    dim3 cfGrid(P / 32, B);                // 128 x 64

    for (int l = 0; l < L; ++l) {
        // x_in = x @ in2f_w[l]
        gemm_bf16_kernel<F, 8, 0, 0><<<gemmGrid, 32, 0, stream>>>(
            x, in2f_t + (size_t)l * F * F, nullptr, x_in);
        // agg = 0
        hipMemsetAsync(agg, 0, (size_t)B * NATOM * F * 4, stream);
        // fused rbf -> filter MLP -> gather*mul -> scatter-add
        cfconv_kernel<<<cfGrid, 32, 0, stream>>>(
            dists, idxj, x_in,
            w1t + (size_t)l * F * NRBF_PAD, fb1 + l * F,
            w2t + (size_t)l * F * F,        fb2 + l * F, agg);
        // tmp1 = ssp(agg @ gw1 + gb1)
        gemm_bf16_kernel<F, 8, 1, 0><<<gemmGrid, 32, 0, stream>>>(
            agg, g1t + (size_t)l * F * F, gb1 + l * F, tmp1);
        // x += tmp1 @ gw2 + gb2
        gemm_bf16_kernel<F, 8, 0, 1><<<gemmGrid, 32, 0, stream>>>(
            tmp1, g2t + (size_t)l * F * F, gb2 + l * F, x);
    }

    head_kernel<<<B, 128, 0, stream>>>(x, ow1, ow2, ob2, out);
}
